// MultiHeadGATLayer_69071664054641
// MI455X (gfx1250) — matrix operand order repaired and checked by reading, not compiled, not run on hardware
//
#include <hip/hip_runtime.h>

#define N_NODES   50000
#define N_EDGES   800000
#define IN_CH     256
#define HEADS     8
#define HEAD_DIM  32
#define OUT_CH    256          // HEADS * HEAD_DIM
#define NEG_SLOPE 0.2f
#define LN_EPS    1e-5f
#define SM_EPS    1e-16f

typedef __attribute__((ext_vector_type(16))) __bf16 v16bf;
typedef __attribute__((ext_vector_type(8)))  float  v8f;
typedef __attribute__((ext_vector_type(4)))  int    v4i;

// ---------------------------------------------------------------- fp32->bf16
__global__ __launch_bounds__(256) void cvt_x_bf16(const float* __restrict__ x,
                                                  unsigned short* __restrict__ xb,
                                                  int n) {
  int i = blockIdx.x * blockDim.x + threadIdx.x;
  if (i < n) {
    __bf16 h = (__bf16)x[i];
    xb[i] = __builtin_bit_cast(unsigned short, h);
  }
}

// lin_w is [IN_CH][OUT_CH] row-major; store bf16 transposed [OUT_CH][IN_CH]
// so WMMA B-fragment loads are contiguous 16B chunks along K.
__global__ __launch_bounds__(256) void cvt_w_bf16_t(const float* __restrict__ w,
                                                    unsigned short* __restrict__ wt) {
  int i = blockIdx.x * blockDim.x + threadIdx.x;   // i = k*OUT_CH + n (coalesced read)
  if (i < IN_CH * OUT_CH) {
    int k = i / OUT_CH, n = i % OUT_CH;
    __bf16 h = (__bf16)w[i];
    wt[(size_t)n * IN_CH + k] = __builtin_bit_cast(unsigned short, h);
  }
}

// ---------------------------------------------------------------- WMMA GEMM
// xp[50000][256] = x_bf16 @ W_bf16, fp32 accumulate.
// One wave32 per 16(M)x64(N) tile: A fragment loaded once per K-step and
// reused by 4 back-to-back v_wmma_f32_16x16x32_bf16 (4x less A traffic,
// back-to-back matrix-pipe issue). 32 WMMAs per wave.
__global__ __launch_bounds__(256) void gat_gemm_wmma(
    const unsigned short* __restrict__ xb,   // A: [N_NODES][IN_CH] bf16 row-major
    const unsigned short* __restrict__ wt,   // B^T: [OUT_CH][IN_CH] bf16
    float* __restrict__ xp)                  // C: [N_NODES][OUT_CH] f32
{
  const int lane = threadIdx.x & 31;
  const int wave = threadIdx.x >> 5;
  const int NT   = OUT_CH / 64;              // 4 N-supertiles of 64
  int tile = blockIdx.x * 8 + wave;
  if (tile >= (N_NODES / 16) * NT) return;
  int m0 = (tile / NT) * 16;
  int n0 = (tile % NT) * 64;

  const int half = lane >> 4;                // 0: lanes 0-15, 1: lanes 16-31
  const int r    = lane & 15;

  // A fragment (16x32 bf16): lanes 0-15 hold row m0+r, K {0..7,16..23};
  // lanes 16-31 hold row m0+r, K {8..15,24..31}.
  const __bf16* A  = (const __bf16*)xb + (size_t)(m0 + r) * IN_CH + (half ? 8 : 0);
  // B fragment (32x16 bf16, column n0+16t+r): lanes 0-15 K {0..7,16..23}, lanes 16-31 K {8..15,24..31}.
  const __bf16* B0 = (const __bf16*)wt + (size_t)(n0 + r) * IN_CH + (half ? 8 : 0);

  v8f acc0 = {0.f,0.f,0.f,0.f,0.f,0.f,0.f,0.f};
  v8f acc1 = acc0, acc2 = acc0, acc3 = acc0;

#pragma unroll
  for (int k0 = 0; k0 < IN_CH; k0 += 32) {
    union { v16bf v; v4i q[2]; } ua, ub0, ub1, ub2, ub3;
    ua.q[0] = *(const v4i*)(A + k0);
    ua.q[1] = *(const v4i*)(A + k0 + 16);
    const __bf16* B = B0 + k0;
    ub0.q[0] = *(const v4i*)(B);
    ub0.q[1] = *(const v4i*)(B + 16);
    ub1.q[0] = *(const v4i*)(B + 16 * IN_CH);
    ub1.q[1] = *(const v4i*)(B + 16 * IN_CH + 16);
    ub2.q[0] = *(const v4i*)(B + 32 * IN_CH);
    ub2.q[1] = *(const v4i*)(B + 32 * IN_CH + 16);
    ub3.q[0] = *(const v4i*)(B + 48 * IN_CH);
    ub3.q[1] = *(const v4i*)(B + 48 * IN_CH + 16);
    if (k0 + 32 < IN_CH) {
      __builtin_prefetch((const void*)(A + k0 + 32), 0, 3);  // global_prefetch_b8
      __builtin_prefetch((const void*)(B + 32), 0, 3);
    }
    acc0 = __builtin_amdgcn_wmma_f32_16x16x32_bf16(false, ua.v, false, ub0.v,
                                                   (short)0, acc0, false, false);
    acc1 = __builtin_amdgcn_wmma_f32_16x16x32_bf16(false, ua.v, false, ub1.v,
                                                   (short)0, acc1, false, false);
    acc2 = __builtin_amdgcn_wmma_f32_16x16x32_bf16(false, ua.v, false, ub2.v,
                                                   (short)0, acc2, false, false);
    acc3 = __builtin_amdgcn_wmma_f32_16x16x32_bf16(false, ua.v, false, ub3.v,
                                                   (short)0, acc3, false, false);
  }

  // C layout per 16x16 sub-tile: VGPR v -> M = v (+8 for lanes 16-31), N = lane&15.
  float* crow = xp + (size_t)(m0 + (half ? 8 : 0)) * OUT_CH + n0 + r;
#pragma unroll
  for (int v = 0; v < 8; ++v) {
    crow[(size_t)v * OUT_CH +  0] = acc0[v];
    crow[(size_t)v * OUT_CH + 16] = acc1[v];
    crow[(size_t)v * OUT_CH + 32] = acc2[v];
    crow[(size_t)v * OUT_CH + 48] = acc3[v];
  }
}

// ------------------------------------------------- per-(node,head) att scores
__global__ __launch_bounds__(256) void gat_att_scores(
    const float* __restrict__ xp, const float* __restrict__ att,
    float* __restrict__ ai, float* __restrict__ aj,
    float* __restrict__ amax, float* __restrict__ denom) {
  int i = blockIdx.x * blockDim.x + threadIdx.x;
  if (i >= N_NODES * HEADS) return;
  int n = i / HEADS, h = i % HEADS;
  const float* row = xp + (size_t)n * OUT_CH + h * HEAD_DIM;
  const float* wi  = att + h * 2 * HEAD_DIM;       // att_i
  const float* wj  = wi + HEAD_DIM;                // att_j
  float si = 0.f, sj = 0.f;
#pragma unroll
  for (int c = 0; c < HEAD_DIM; ++c) { float v = row[c]; si += v * wi[c]; sj += v * wj[c]; }
  ai[i] = si; aj[i] = sj;
  amax[i]  = -__builtin_huge_valf();
  denom[i] = 0.f;
}

// ------------------------------------------------- float atomic max (bit trick)
__device__ __forceinline__ void atomicMaxF(float* addr, float val) {
  if (val >= 0.f) atomicMax((int*)addr, __float_as_int(val));
  else            atomicMin((unsigned int*)addr, (unsigned int)__float_as_int(val));
}

// ------------------------------------------------- edge: leaky-relu + seg max
__global__ __launch_bounds__(256) void gat_edge_alpha(
    const int* __restrict__ src, const int* __restrict__ dst,
    const float* __restrict__ ai, const float* __restrict__ aj,
    float* __restrict__ exbuf, float* __restrict__ amax) {
  int i = blockIdx.x * blockDim.x + threadIdx.x;
  if (i >= N_EDGES * HEADS) return;
  int e = i / HEADS, h = i % HEADS;
  int s = src[e], d = dst[e];
  float al = ai[d * HEADS + h] + aj[s * HEADS + h];
  al = (al >= 0.f) ? al : NEG_SLOPE * al;
  exbuf[i] = al;
  atomicMaxF(&amax[d * HEADS + h], al);
}

// ------------------------------------------------- edge: exp + seg sum
__global__ __launch_bounds__(256) void gat_edge_exp(
    const int* __restrict__ dst, const float* __restrict__ amax,
    float* __restrict__ exbuf, float* __restrict__ denom) {
  int i = blockIdx.x * blockDim.x + threadIdx.x;
  if (i >= N_EDGES * HEADS) return;
  int e = i / HEADS, h = i % HEADS;
  int d = dst[e];
  float m = amax[d * HEADS + h];
  if (!(m > -1e37f && m < 1e37f)) m = 0.f;         // amax = 0 where not finite
  float ex = __expf(exbuf[i] - m);
  exbuf[i] = ex;
  atomicAdd(&denom[d * HEADS + h], ex);
}

// ------------------------------------------------- edge: weighted scatter-add
// One wave32 per edge; lanes 0-7 compute the per-head weights once, then
// broadcast via __shfl; lane handles channel (h*32 + lane) for each head h.
__global__ __launch_bounds__(256) void gat_aggregate(
    const int* __restrict__ src, const int* __restrict__ dst,
    const float* __restrict__ xp, const float* __restrict__ exbuf,
    const float* __restrict__ denom, float* __restrict__ accum) {
  int e = blockIdx.x * 8 + (threadIdx.x >> 5);
  if (e >= N_EDGES) return;
  int lane = threadIdx.x & 31;
  int s = src[e], d = dst[e];
  const float* xr   = xp + (size_t)s * OUT_CH;
  float*       orow = accum + (size_t)d * OUT_CH;
  float wlocal = 0.f;
  if (lane < HEADS)
    wlocal = exbuf[e * HEADS + lane] / (denom[d * HEADS + lane] + SM_EPS);
#pragma unroll
  for (int h = 0; h < HEADS; ++h) {
    float w = __shfl(wlocal, h, 32);
    int c = h * HEAD_DIM + lane;
    atomicAdd(&orow[c], w * xr[c]);
  }
}

// ------------------------------------------------- LayerNorm + ELU + residual
// One wave32 per node; 8 channels per lane; shfl_xor tree reduction (wave32).
__global__ __launch_bounds__(256) void gat_ln_elu_res(
    float* __restrict__ out, const float* __restrict__ x,
    const float* __restrict__ lnw, const float* __restrict__ lnb) {
  int n = blockIdx.x * 8 + (threadIdx.x >> 5);
  if (n >= N_NODES) return;
  int lane = threadIdx.x & 31;
  float v[8];
  float s = 0.f;
#pragma unroll
  for (int k = 0; k < 8; ++k) { v[k] = out[(size_t)n * OUT_CH + k * 32 + lane]; s += v[k]; }
#pragma unroll
  for (int m = 16; m >= 1; m >>= 1) s += __shfl_xor(s, m, 32);
  float mu = s * (1.f / OUT_CH);
  float var = 0.f;
#pragma unroll
  for (int k = 0; k < 8; ++k) { float dlt = v[k] - mu; var += dlt * dlt; }
#pragma unroll
  for (int m = 16; m >= 1; m >>= 1) var += __shfl_xor(var, m, 32);
  float inv = rsqrtf(var * (1.f / OUT_CH) + LN_EPS);
#pragma unroll
  for (int k = 0; k < 8; ++k) {
    int c = k * 32 + lane;
    float y = (v[k] - mu) * inv * lnw[c] + lnb[c];
    y = (y > 0.f) ? y : (__expf(y) - 1.f);          // ELU
    out[(size_t)n * OUT_CH + c] = y + x[(size_t)n * OUT_CH + c];
  }
}

// ---------------------------------------------------------------- launcher
extern "C" void kernel_launch(void* const* d_in, const int* in_sizes, int n_in,
                              void* d_out, int out_size, void* d_ws, size_t ws_size,
                              hipStream_t stream) {
  const float* x    = (const float*)d_in[0];
  const int*   ei   = (const int*)  d_in[1];   // [2][N_EDGES]
  const float* linw = (const float*)d_in[2];
  const float* att  = (const float*)d_in[3];
  const float* lnw  = (const float*)d_in[4];
  const float* lnb  = (const float*)d_in[5];
  const int* src = ei;
  const int* dst = ei + N_EDGES;

  // workspace carve-up (all offsets 256B-aligned); total ~109 MB
  char* ws = (char*)d_ws;
  unsigned short* xb = (unsigned short*)ws; ws += (size_t)N_NODES * IN_CH * 2;
  unsigned short* wt = (unsigned short*)ws; ws += (size_t)IN_CH * OUT_CH * 2;
  float* xp   = (float*)ws; ws += (size_t)N_NODES * OUT_CH * 4;
  float* ai   = (float*)ws; ws += (size_t)N_NODES * HEADS * 4;
  float* aj   = (float*)ws; ws += (size_t)N_NODES * HEADS * 4;
  float* amax = (float*)ws; ws += (size_t)N_NODES * HEADS * 4;
  float* den  = (float*)ws; ws += (size_t)N_NODES * HEADS * 4;
  float* exb  = (float*)ws;

  float* accum = (float*)d_out;                     // aggregate in-place in d_out
  hipMemsetAsync(accum, 0, (size_t)N_NODES * OUT_CH * sizeof(float), stream);

  { int n = N_NODES * IN_CH;
    cvt_x_bf16<<<(n + 255) / 256, 256, 0, stream>>>(x, xb, n); }
  cvt_w_bf16_t<<<(IN_CH * OUT_CH + 255) / 256, 256, 0, stream>>>(linw, wt);

  { int tiles = (N_NODES / 16) * (OUT_CH / 64);     // 12500 wave-tiles
    gat_gemm_wmma<<<(tiles + 7) / 8, 256, 0, stream>>>(xb, wt, xp); }

  gat_att_scores<<<(N_NODES * HEADS + 255) / 256, 256, 0, stream>>>(xp, att, ai, aj, amax, den);
  gat_edge_alpha<<<(N_EDGES * HEADS + 255) / 256, 256, 0, stream>>>(src, dst, ai, aj, exb, amax);
  gat_edge_exp  <<<(N_EDGES * HEADS + 255) / 256, 256, 0, stream>>>(dst, amax, exb, den);
  gat_aggregate <<<(N_EDGES + 7) / 8, 256, 0, stream>>>(src, dst, xp, exb, den, accum);
  gat_ln_elu_res<<<(N_NODES + 7) / 8, 256, 0, stream>>>(accum, x, lnw, lnb);
}
